// SDGCN_24283745091813
// MI455X (gfx1250) — matrix operand order repaired and checked by reading, not compiled, not run on hardware
//
#include <hip/hip_runtime.h>

// ---------------------------------------------------------------------------
// SDGCN on MI455X (gfx1250): WMMA f16 GEMMs (pre-swizzled weights, vectorized
// unpredicated A loads, per-tile B prefetch) + float4 atomic edge scatters.
// HID = 32 == wave32 width -> lane == channel for all node-wise kernels.
// ---------------------------------------------------------------------------

typedef __attribute__((ext_vector_type(16))) _Float16 v16h;
typedef __attribute__((ext_vector_type(8)))  float    v8f;

#define TB 256  // generic block size

__device__ __forceinline__ float wave_sum32(float x) {
#pragma unroll
  for (int off = 16; off > 0; off >>= 1) x += __shfl_xor(x, off, 32);
  return x;
}

// ------------------- weight pre-swizzle into WMMA B layout -----------------
// B 32x16 f16 fragment layout per k-step/tile: lane holds column nt*16+(lane&15),
// halves i hold k = ks*32 + (lane>=16?8:0) + (i<8 ? i : 8+i).
// Output: Bsw[((nt*KS + ks)*32 + lane)*16 + i]  (contiguous 32B per lane).
__global__ void k_swz_w(const float* __restrict__ W, _Float16* __restrict__ Bsw,
                        int K, int Nout) {
  const int NT = (Nout + 15) >> 4, KS = K >> 5;
  const int total = NT * KS * 512;
  int i = blockIdx.x * blockDim.x + threadIdx.x;
  if (i >= total) return;
  const int e    = i & 15;
  const int lane = (i >> 4) & 31;
  const int ks   = (i >> 9) % KS;
  const int nt   = (i >> 9) / KS;
  const int v = e >> 1, hlf = e & 1;
  const int kb = ks * 32 + ((lane >> 4) ? 8 : 0) + ((e < 8) ? e : 8 + e);
  const int nB = nt * 16 + (lane & 15);
  Bsw[i] = (nB < Nout) ? (_Float16)W[(long)kb * Nout + nB] : (_Float16)0.f;
}

// ------------------------- generic WMMA GEMM -------------------------------
// out[n x NOUT] = A[n x K] @ W  (W pre-swizzled f16).  One wave = 16-row stripe.
// A 16x32 f16 layout: lane row = mbase+(lane&15); halves 0..7 <- k0+laneHi*8+0..7,
// halves 8..15 <- k0+16+laneHi*8+0..7  (two contiguous float4-pairs -> b128s).
template <int K, int NOUT>
__global__ void k_gemm_wmma(const float* __restrict__ A,
                            const _Float16* __restrict__ Bsw,
                            float* __restrict__ out, int n) {
  constexpr int KS = K / 32;
  constexpr int NT = (NOUT + 15) / 16;
  const int lane   = threadIdx.x & 31;
  const int wave   = threadIdx.x >> 5;
  const int mbase  = (blockIdx.x * (blockDim.x >> 5) + wave) * 16;
  const int lo     = lane & 15;
  const int laneHi = lane >> 4;
  int mA = mbase + lo;
  if (mA >= n) mA = n - 1;                 // clamp: loads always legal, tail
  const float* __restrict__ Arow = A + (long)mA * K;  // rows' stores are masked

  // Load & convert all A fragments for this row stripe (K <= 256 -> <= 8 frags)
  v16h afrag[KS];
#pragma unroll
  for (int ks = 0; ks < KS; ++ks) {
    const int kb = ks * 32 + laneHi * 8;
    const float4 f0 = *reinterpret_cast<const float4*>(Arow + kb);
    const float4 f1 = *reinterpret_cast<const float4*>(Arow + kb + 4);
    const float4 f2 = *reinterpret_cast<const float4*>(Arow + kb + 16);
    const float4 f3 = *reinterpret_cast<const float4*>(Arow + kb + 20);
    v16h a;
    a[0]  = (_Float16)f0.x; a[1]  = (_Float16)f0.y;
    a[2]  = (_Float16)f0.z; a[3]  = (_Float16)f0.w;
    a[4]  = (_Float16)f1.x; a[5]  = (_Float16)f1.y;
    a[6]  = (_Float16)f1.z; a[7]  = (_Float16)f1.w;
    a[8]  = (_Float16)f2.x; a[9]  = (_Float16)f2.y;
    a[10] = (_Float16)f2.z; a[11] = (_Float16)f2.w;
    a[12] = (_Float16)f3.x; a[13] = (_Float16)f3.y;
    a[14] = (_Float16)f3.z; a[15] = (_Float16)f3.w;
    afrag[ks] = a;
  }

#pragma unroll
  for (int nt = 0; nt < NT; ++nt) {
    // Prefetch all B fragments for this tile -> one load clause, partial waits
    v16h bfrag[KS];
#pragma unroll
    for (int ks = 0; ks < KS; ++ks)
      bfrag[ks] = *reinterpret_cast<const v16h*>(
          Bsw + (((long)nt * KS + ks) * 32 + lane) * 16);

    v8f c = {};
#pragma unroll
    for (int ks = 0; ks < KS; ++ks)
      c = __builtin_amdgcn_wmma_f32_16x16x32_f16(false, afrag[ks], false,
                                                 bfrag[ks], (short)0, c,
                                                 false, false);

    // f32 C/D 16x16: lane -> column lane&15; VGPR r -> row r (+8 for hi lanes)
    const int colBase = nt * 16;
    if (mbase + 16 <= n && colBase + 16 <= NOUT) {
#pragma unroll
      for (int r = 0; r < 8; ++r)
        out[(long)(mbase + r + laneHi * 8) * NOUT + colBase + lo] = c[r];
    } else {
#pragma unroll
      for (int r = 0; r < 8; ++r) {
        const int m = mbase + r + laneHi * 8;
        const int col = colBase + lo;
        if (m < n && col < NOUT) out[(long)m * NOUT + col] = c[r];
      }
    }
  }
}

// ----------------------------- small utilities -----------------------------
__global__ void k_zero(float* __restrict__ p, long n) {
  long i = (long)blockIdx.x * blockDim.x + threadIdx.x;
  if (i < n) p[i] = 0.f;
}

__global__ void k_degcnt(const int* __restrict__ row, const int* __restrict__ col,
                         float* __restrict__ deg, float* __restrict__ cnt, int ne) {
  int e = blockIdx.x * blockDim.x + threadIdx.x;
  if (e < ne) {
    atomicAdd(&deg[col[e]], 1.f);
    atomicAdd(&cnt[row[e]], 1.f);
  }
}

__global__ void k_dinv(const float* __restrict__ deg, float* __restrict__ cnt,
                       float* __restrict__ dinv, int n) {
  int i = blockIdx.x * blockDim.x + threadIdx.x;
  if (i < n) {
    dinv[i] = rsqrtf(deg[i] + 1.f);      // +1 self loop; always > 0
    cnt[i]  = fmaxf(cnt[i], 1.f);
  }
}

__global__ void k_enorm(const int* __restrict__ row, const int* __restrict__ col,
                        const float* __restrict__ dinv, float* __restrict__ en, int ne) {
  int e = blockIdx.x * blockDim.x + threadIdx.x;
  if (e < ne) en[e] = dinv[row[e]] * dinv[col[e]];
}

// ------------------------------ edge scatters ------------------------------
// agg[col] += t[row] * enorm   (C4 float4-chunks per edge)
__global__ void k_scatter_gcn(const float* __restrict__ t, const float* __restrict__ en,
                              const int* __restrict__ row, const int* __restrict__ col,
                              float* __restrict__ agg, int ne, int C4) {
  long gid = (long)blockIdx.x * blockDim.x + threadIdx.x;
  int e = (int)(gid / C4);
  if (e >= ne) return;
  const int c = (int)(gid % C4) * 4;
  const int C = C4 * 4;
  const float4 v = *reinterpret_cast<const float4*>(&t[(long)row[e] * C + c]);
  const float s = en[e];
  float* dst = &agg[(long)col[e] * C + c];
  atomicAdd(dst + 0, v.x * s);
  atomicAdd(dst + 1, v.y * s);
  atomicAdd(dst + 2, v.z * s);
  atomicAdd(dst + 3, v.w * s);
}

// hmsum[row] += h[col]   (C = 32)
__global__ void k_scatter_mean(const float* __restrict__ h,
                               const int* __restrict__ row, const int* __restrict__ col,
                               float* __restrict__ hmsum, int ne) {
  long gid = (long)blockIdx.x * blockDim.x + threadIdx.x;
  int e = (int)(gid >> 3);
  if (e >= ne) return;
  const int c = (int)(gid & 7) * 4;
  const float4 v = *reinterpret_cast<const float4*>(&h[(long)col[e] * 32 + c]);
  float* dst = &hmsum[(long)row[e] * 32 + c];
  atomicAdd(dst + 0, v.x);
  atomicAdd(dst + 1, v.y);
  atomicAdd(dst + 2, v.z);
  atomicAdd(dst + 3, v.w);
}

// hstdsum[row] += |h[col] - hmsum[col]/cnt[col]|   (C = 32)
__global__ void k_scatter_mad(const float* __restrict__ h, const float* __restrict__ hmsum,
                              const float* __restrict__ cnt,
                              const int* __restrict__ row, const int* __restrict__ col,
                              float* __restrict__ hstdsum, int ne) {
  long gid = (long)blockIdx.x * blockDim.x + threadIdx.x;
  int e = (int)(gid >> 3);
  if (e >= ne) return;
  const int c  = (int)(gid & 7) * 4;
  const int cl = col[e];
  const float inv = 1.f / cnt[cl];
  const float4 hv = *reinterpret_cast<const float4*>(&h[(long)cl * 32 + c]);
  const float4 hm = *reinterpret_cast<const float4*>(&hmsum[(long)cl * 32 + c]);
  float* dst = &hstdsum[(long)row[e] * 32 + c];
  atomicAdd(dst + 0, fabsf(hv.x - hm.x * inv));
  atomicAdd(dst + 1, fabsf(hv.y - hm.y * inv));
  atomicAdd(dst + 2, fabsf(hv.z - hm.z * inv));
  atomicAdd(dst + 3, fabsf(hv.w - hm.w * inv));
}

// ---------------------------- node-wise kernels ----------------------------
// h = LN(relu(agg + t*dinv^2 + b)) ; wave per node, lane == channel (HID=32)
__global__ void k_node_gcnpost(const float* __restrict__ agg, const float* __restrict__ t,
                               const float* __restrict__ dinv, const float* __restrict__ b,
                               const float* __restrict__ g, const float* __restrict__ bb,
                               float* __restrict__ h, int n) {
  const int node = blockIdx.x * (blockDim.x >> 5) + (threadIdx.x >> 5);
  const int lane = threadIdx.x & 31;
  if (node >= n) return;                       // uniform per wave
  const long idx = (long)node * 32 + lane;
  const float di = dinv[node];
  float u = agg[idx] + t[idx] * di * di + b[lane];
  u = fmaxf(u, 0.f);
  const float mean = wave_sum32(u) * (1.f / 32.f);
  const float d = u - mean;
  const float var = wave_sum32(d * d) * (1.f / 32.f);
  h[idx] = d * rsqrtf(var + 1e-5f) * g[lane] + bb[lane];
}

// x0 = LN(relu(t)) ; no bias
__global__ void k_relu_ln(const float* __restrict__ t, const float* __restrict__ g,
                          const float* __restrict__ bb, float* __restrict__ o, int n) {
  const int node = blockIdx.x * (blockDim.x >> 5) + (threadIdx.x >> 5);
  const int lane = threadIdx.x & 31;
  if (node >= n) return;
  const long idx = (long)node * 32 + lane;
  float u = fmaxf(t[idx], 0.f);
  const float mean = wave_sum32(u) * (1.f / 32.f);
  const float d = u - mean;
  const float var = wave_sum32(d * d) * (1.f / 32.f);
  o[idx] = d * rsqrtf(var + 1e-5f) * g[lane] + bb[lane];
}

// score gate + residual mix (in place on h)
__global__ void k_node_mix(float* __restrict__ h, const float* __restrict__ hmsum,
                           const float* __restrict__ hstdsum, const float* __restrict__ cnt,
                           const float* __restrict__ w_ws, const float* __restrict__ x0,
                           int n) {
  const int node = blockIdx.x * (blockDim.x >> 5) + (threadIdx.x >> 5);
  const int lane = threadIdx.x & 31;
  if (node >= n) return;
  const long idx = (long)node * 32 + lane;
  const float inv = 1.f / cnt[node];
  const float hm = hmsum[idx] * inv;
  const float hs = hstdsum[idx] * inv;
  const float hv = h[idx];
  float p = hm * hv * w_ws[lane] + hs * w_ws[32 + lane] + hv * w_ws[64 + lane];
  float s = wave_sum32(p);
  s = 1.f / (1.f + __expf(-s));
  h[idx] = (1.f - s) * hv + s * x0[idx];
}

// out = agg + t*dinv^2 + bl   (final GCN, C = NCLS)
__global__ void k_final(const float* __restrict__ agg, const float* __restrict__ t,
                        const float* __restrict__ dinv, const float* __restrict__ bl,
                        float* __restrict__ out, int n, int C) {
  long gid = (long)blockIdx.x * blockDim.x + threadIdx.x;
  if (gid >= (long)n * C) return;
  const int node = (int)(gid / C);
  const int c    = (int)(gid % C);
  const float di = dinv[node];
  out[gid] = agg[gid] + t[gid] * di * di + bl[c];
}

// ---------------------------------------------------------------------------
static inline int cdivl(long a, long b) { return (int)((a + b - 1) / b); }

extern "C" void kernel_launch(void* const* d_in, const int* in_sizes, int n_in,
                              void* d_out, int out_size, void* d_ws, size_t ws_size,
                              hipStream_t stream) {
  const float* x    = (const float*)d_in[0];
  const int*   ei   = (const int*)  d_in[1];
  const float* W0   = (const float*)d_in[2];
  const float* b0   = (const float*)d_in[3];
  const float* Whh  = (const float*)d_in[4];
  const float* bhh  = (const float*)d_in[5];
  const float* Wl   = (const float*)d_in[6];
  const float* bl   = (const float*)d_in[7];
  const float* lng0 = (const float*)d_in[8];
  const float* lnb0 = (const float*)d_in[9];
  const float* lngm = (const float*)d_in[10];
  const float* lnbm = (const float*)d_in[11];
  const float* w_ws = (const float*)d_in[12];
  const float* Wres = (const float*)d_in[13];
  const float* rg   = (const float*)d_in[14];
  const float* rb   = (const float*)d_in[15];

  const int n    = in_sizes[0] / 256;     // 50000
  const int ne   = in_sizes[1] / 2;       // 1600000
  const int ncls = in_sizes[7];           // 40
  const int nmid = in_sizes[5] / 32;      // 30
  const int* row = ei;
  const int* col = ei + ne;
  const int tmax = (ncls > 32) ? ncls : 32;

  float* ws    = (float*)d_ws;
  float* dinv  = ws;                       // n
  float* cnt   = ws + n;                   // n
  float* enorm = ws + 2L * n;              // ne
  float* t     = enorm + ne;               // n * tmax
  float* agg   = t + (long)n * tmax;       // n * tmax  (reused as hmsum)
  float* h     = agg + (long)n * tmax;     // n * 32
  float* x0    = h + (long)n * 32;         // n * 32
  // (t is reused as hstdsum between node kernels)

  // swizzled f16 weight buffers
  _Float16* fw      = (_Float16*)(x0 + (long)n * 32);
  const int SZ_256_32 = 2 * 8 * 512;              // NT=2, KS=8
  const int SZ_32_32  = 2 * 1 * 512;              // NT=2, KS=1
  const int SZ_32_CLS = ((ncls + 15) >> 4) * 512; // NT, KS=1
  _Float16* Wres_sw = fw;
  _Float16* W0_sw   = Wres_sw + SZ_256_32;
  _Float16* Whh_sw  = W0_sw + SZ_256_32;          // nmid consecutive blocks
  _Float16* Wl_sw   = Whh_sw + (long)nmid * SZ_32_32;

  // ---- weight pre-swizzle (once per call) ----
  k_swz_w<<<cdivl(SZ_256_32, TB), TB, 0, stream>>>(Wres, Wres_sw, 256, 32);
  k_swz_w<<<cdivl(SZ_256_32, TB), TB, 0, stream>>>(W0, W0_sw, 256, 32);
  for (int l = 0; l < nmid; ++l)
    k_swz_w<<<cdivl(SZ_32_32, TB), TB, 0, stream>>>(Whh + (long)l * 32 * 32,
                                                    Whh_sw + (long)l * SZ_32_32, 32, 32);
  k_swz_w<<<cdivl(SZ_32_CLS, TB), TB, 0, stream>>>(Wl, Wl_sw, 32, ncls);

  // ---- graph normalization precompute ----
  k_zero  <<<cdivl(n, TB), TB, 0, stream>>>(t, n);          // deg scratch
  k_zero  <<<cdivl(n, TB), TB, 0, stream>>>(cnt, n);
  k_degcnt<<<cdivl(ne, TB), TB, 0, stream>>>(row, col, t, cnt, ne);
  k_dinv  <<<cdivl(n, TB), TB, 0, stream>>>(t, cnt, dinv, n);
  k_enorm <<<cdivl(ne, TB), TB, 0, stream>>>(row, col, dinv, enorm, ne);

  // ---- residual path: x0 = LN(relu(x @ W_res)) ----
  k_gemm_wmma<256, 32><<<cdivl(n, 128), 256, 0, stream>>>(x, Wres_sw, t, n);
  k_relu_ln<<<cdivl(n, 8), 256, 0, stream>>>(t, rg, rb, x0, n);

  // ---- one GCN block (post-GEMM part) ----
  auto run_block_rest = [&](const float* b, const float* g, const float* bb) {
    k_zero        <<<cdivl((long)n * 32, TB), TB, 0, stream>>>(agg, (long)n * 32);
    k_scatter_gcn <<<cdivl((long)ne * 8, TB), TB, 0, stream>>>(t, enorm, row, col, agg, ne, 8);
    k_node_gcnpost<<<cdivl(n, 8), 256, 0, stream>>>(agg, t, dinv, b, g, bb, h, n);
    k_zero        <<<cdivl((long)n * 32, TB), TB, 0, stream>>>(agg, (long)n * 32); // hmsum
    k_zero        <<<cdivl((long)n * 32, TB), TB, 0, stream>>>(t,   (long)n * 32); // hstdsum
    k_scatter_mean<<<cdivl((long)ne * 8, TB), TB, 0, stream>>>(h, row, col, agg, ne);
    k_scatter_mad <<<cdivl((long)ne * 8, TB), TB, 0, stream>>>(h, agg, cnt, row, col, t, ne);
    k_node_mix    <<<cdivl(n, 8), 256, 0, stream>>>(h, agg, t, cnt, w_ws, x0, n);
  };

  // block 0: x @ W0 (K = 256)
  k_gemm_wmma<256, 32><<<cdivl(n, 128), 256, 0, stream>>>(x, W0_sw, t, n);
  run_block_rest(b0, lng0, lnb0);

  // middle blocks: h @ Whh[l] (K = 32)
  for (int l = 0; l < nmid; ++l) {
    k_gemm_wmma<32, 32><<<cdivl(n, 128), 256, 0, stream>>>(h, Whh_sw + (long)l * SZ_32_32,
                                                           t, n);
    run_block_rest(bhh + (long)l * 32, lngm + (long)l * 32, lnbm + (long)l * 32);
  }

  // ---- final GCN 32 -> NCLS ----
  k_gemm_wmma<32, 40><<<cdivl(n, 128), 256, 0, stream>>>(h, Wl_sw, t, n);
  k_zero       <<<cdivl((long)n * ncls, TB), TB, 0, stream>>>(agg, (long)n * ncls);
  k_scatter_gcn<<<cdivl((long)ne * (ncls / 4), TB), TB, 0, stream>>>(t, enorm, row, col,
                                                                     agg, ne, ncls / 4);
  k_final      <<<cdivl((long)n * ncls, TB), TB, 0, stream>>>(agg, t, dinv, bl,
                                                              (float*)d_out, n, ncls);
}
